// FusionRefiner_11141145166145
// MI455X (gfx1250) — matrix operand order repaired and checked by reading, not compiled, not run on hardware
//
#include <hip/hip_runtime.h>
#include <hip/hip_bf16.h>
#include <math.h>

// Problem constants (match reference)
static constexpr int NN = 8192;    // nodes
static constexpr int HH = 256;     // hidden dim
static constexpr int EE = 262144;  // edges
static constexpr int VV = 2;       // views

typedef __attribute__((ext_vector_type(16))) _Float16 v16h;
typedef __attribute__((ext_vector_type(8)))  _Float16 v8h;
typedef __attribute__((ext_vector_type(8)))  float    v8f;

// ---------------------------------------------------------------------------
// small utility kernels
// ---------------------------------------------------------------------------
__global__ void k_deg_init(float* deg, int n) {
    int i = blockIdx.x * blockDim.x + threadIdx.x;
    if (i < n) deg[i] = 1.0f;   // self-loop contribution
}

__global__ void k_deg_count(const int* dst, float* deg, int e) {
    for (int i = blockIdx.x * blockDim.x + threadIdx.x; i < e;
         i += gridDim.x * blockDim.x) {
        atomicAdd(&deg[dst[i]], 1.0f);
    }
}

__global__ void k_dinv(const float* deg, float* dinv, int n) {
    int i = blockIdx.x * blockDim.x + threadIdx.x;
    if (i < n) dinv[i] = rsqrtf(deg[i]);
}

__global__ void k_cvt_f32_f16(const float* in, _Float16* out, int n) {
    for (int i = blockIdx.x * blockDim.x + threadIdx.x; i < n;
         i += gridDim.x * blockDim.x) {
        out[i] = (_Float16)in[i];
    }
}

// out[n*rows + k] = (f16) in[k*cols + n]   -- transpose+convert (for weights)
__global__ void k_cvt_transpose(const float* in, _Float16* out, int rows, int cols) {
    int i = blockIdx.x * blockDim.x + threadIdx.x;
    if (i < rows * cols) {
        int k = i / cols;
        int n = i - k * cols;
        out[(size_t)n * rows + k] = (_Float16)in[i];
    }
}

// out[i,:] = h[i,:] * dinv[i]^2   (self-loop term of symmetric-normalized A+I)
__global__ void k_agg_init(const float* h, const float* dinv, float* out, int nh) {
    for (int i = blockIdx.x * blockDim.x + threadIdx.x; i < nh;
         i += gridDim.x * blockDim.x) {
        int row = i >> 8;                       // H == 256
        float di = dinv[row];
        out[i] = h[i] * di * di;
    }
}

// scatter-add over edges: out[dst] += h[src] * dinv[src]*dinv[dst]
__global__ void k_agg_edges(const float* __restrict__ h,
                            const float* __restrict__ dinv,
                            const int* __restrict__ src,
                            const int* __restrict__ dst,
                            float* out, int e) {
    const long total = (long)e * (HH / 4);      // 4 features per thread
    for (long i = blockIdx.x * (long)blockDim.x + threadIdx.x; i < total;
         i += gridDim.x * (long)blockDim.x) {
        int ed = (int)(i >> 6);                 // H/4 == 64
        int f  = ((int)i & 63) << 2;
        int s = src[ed], d = dst[ed];
        float nrm = dinv[s] * dinv[d];
        const float* hp = h + (size_t)s * HH + f;
        float*       op = out + (size_t)d * HH + f;
        atomicAdd(op + 0, hp[0] * nrm);
        atomicAdd(op + 1, hp[1] * nrm);
        atomicAdd(op + 2, hp[2] * nrm);
        atomicAdd(op + 3, hp[3] * nrm);
    }
}

// z = (agg + b)  [optionally ReLU]; write f16 copy (and optional f32 copy)
__global__ void k_finalize(const float* agg, const float* bias,
                           float* f32out, _Float16* f16out,
                           int nh, int do_relu) {
    for (int i = blockIdx.x * blockDim.x + threadIdx.x; i < nh;
         i += gridDim.x * blockDim.x) {
        float v = agg[i] + bias[i & (HH - 1)];
        if (do_relu) v = fmaxf(v, 0.0f);
        if (f32out) f32out[i] = v;
        f16out[i] = (_Float16)v;
    }
}

// ---------------------------------------------------------------------------
// WMMA GEMM (A @ B^T):  C[M,Nc] = A[M,K] @ B[Nc,K]^T   (f16 in, f32 out)
//   Both operands are row-major with contiguous K -> identical 16B-chunk loads.
//   SIG: fuse sigmoid into the epilogue.
// Block tile 128x128, BK=32, 256 threads = 8 waves as 4(m) x 2(n); each wave
// owns a 32x64 tile via 2x4 v_wmma_f32_16x16x32_f16 accumulators.
// Double-buffered LDS with register staging: tile k+1 is global_load_b128'd
// while tile k computes; staged registers are ds_store_b128'd after compute.
// ---------------------------------------------------------------------------
template <bool SIG>
__global__ __launch_bounds__(256)
void k_gemm_bt_wmma(const _Float16* __restrict__ A,
                    const _Float16* __restrict__ B,
                    float* __restrict__ C,
                    int M, int Nc, int K) {
    constexpr int BM = 128, BN = 128, BK = 32;
    __shared__ __align__(16) _Float16 sA[2][BM * BK];   // [m][k]
    __shared__ __align__(16) _Float16 sB[2][BN * BK];   // [n][k]

    const int tid  = threadIdx.x;
    const int wave = tid >> 5;
    const int lane = tid & 31;
    const int mo = (wave >> 1) * 32;    // 4 wave-rows
    const int no = (wave & 1) * 64;     // 2 wave-cols

    const int blockN = blockIdx.x * BN;
    const int blockM = blockIdx.y * BM;

    // 16B chunks: BM*BK/8 = 512 chunks, 2 per thread (same geometry for B)
    const int r0 = tid >> 2,         c0 = (tid & 3) * 8;
    const int r1 = (tid + 256) >> 2, c1 = ((tid + 256) & 3) * 8;
    const size_t aro0 = (size_t)(blockM + r0) * K + c0;
    const size_t aro1 = (size_t)(blockM + r1) * K + c1;
    const size_t bro0 = (size_t)(blockN + r0) * K + c0;
    const size_t bro1 = (size_t)(blockN + r1) * K + c1;

    // per-lane WMMA operand addressing (ISA 16-bit A 16x32 / B 32x16 layout)
    const int arow = lane & 15;
    const int kbA  = (lane < 16) ? 0 : 8;    // A lanes 0-15: K=0..7 & 16..23
    const int bcol = lane & 15;
    const int kbB  = (lane < 16) ? 0 : 8;    // B lanes 0-15: K=0..7 & 16..23

    v8f acc[2][4] = {};
    uint4 ra0, ra1, rb0, rb1;

    // ---- prologue: stage tile 0 ----
    ra0 = *(const uint4*)&A[aro0];
    ra1 = *(const uint4*)&A[aro1];
    rb0 = *(const uint4*)&B[bro0];
    rb1 = *(const uint4*)&B[bro1];
    *(uint4*)&sA[0][r0 * BK + c0] = ra0;
    *(uint4*)&sA[0][r1 * BK + c1] = ra1;
    *(uint4*)&sB[0][r0 * BK + c0] = rb0;
    *(uint4*)&sB[0][r1 * BK + c1] = rb1;
    __syncthreads();

    const int NT = K / BK;
    for (int kt = 0; kt < NT; ++kt) {
        const int cur = kt & 1, nxt = cur ^ 1;
        const int k1 = (kt + 1) * BK;

        // issue global loads for the next tile (overlap with compute below)
        if (kt + 1 < NT) {
            ra0 = *(const uint4*)&A[aro0 + k1];
            ra1 = *(const uint4*)&A[aro1 + k1];
            rb0 = *(const uint4*)&B[bro0 + k1];
            rb1 = *(const uint4*)&B[bro1 + k1];
            if (kt + 2 < NT) {   // global_prefetch_b8 for K+2 tile
                __builtin_prefetch(&A[aro0 + k1 + BK], 0, 0);
                __builtin_prefetch(&B[bro0 + k1 + BK], 0, 0);
            }
        }

        // ---- wave compute: 2x4 tiles of 16x16, K=32 per wmma ----
        v16h a[2], b[4];
        #pragma unroll
        for (int ms = 0; ms < 2; ++ms) {
            const _Float16* p = &sA[cur][(mo + ms * 16 + arow) * BK + kbA];
            v8h lo = *(const v8h*)(p);
            v8h hi = *(const v8h*)(p + 16);
            a[ms] = __builtin_shufflevector(lo, hi, 0, 1, 2, 3, 4, 5, 6, 7,
                                            8, 9, 10, 11, 12, 13, 14, 15);
        }
        #pragma unroll
        for (int ns = 0; ns < 4; ++ns) {
            const _Float16* p = &sB[cur][(no + ns * 16 + bcol) * BK + kbB];
            v8h lo = *(const v8h*)(p);
            v8h hi = *(const v8h*)(p + 16);
            b[ns] = __builtin_shufflevector(lo, hi, 0, 1, 2, 3, 4, 5, 6, 7,
                                            8, 9, 10, 11, 12, 13, 14, 15);
        }
        #pragma unroll
        for (int ms = 0; ms < 2; ++ms)
            #pragma unroll
            for (int ns = 0; ns < 4; ++ns)
                acc[ms][ns] = __builtin_amdgcn_wmma_f32_16x16x32_f16(
                    false, a[ms], false, b[ns], (short)0, acc[ms][ns],
                    false, false);

        // publish next tile to the other LDS buffer
        if (kt + 1 < NT) {
            *(uint4*)&sA[nxt][r0 * BK + c0] = ra0;
            *(uint4*)&sA[nxt][r1 * BK + c1] = ra1;
            *(uint4*)&sB[nxt][r0 * BK + c0] = rb0;
            *(uint4*)&sB[nxt][r1 * BK + c1] = rb1;
        }
        __syncthreads();
    }

    // ---- epilogue: C/D layout -> lanes 0-15 rows 0..7, lanes 16-31 rows 8..15
    const int nloc  = lane & 15;
    const int rbase = (lane >> 4) * 8;
    #pragma unroll
    for (int ms = 0; ms < 2; ++ms) {
        #pragma unroll
        for (int ns = 0; ns < 4; ++ns) {
            #pragma unroll
            for (int r = 0; r < 8; ++r) {
                int row = blockM + mo + ms * 16 + rbase + r;
                int col = blockN + no + ns * 16 + nloc;
                float v = acc[ms][ns][r];
                if constexpr (SIG) v = 1.0f / (1.0f + __expf(-v));
                C[(size_t)row * Nc + col] = v;
            }
        }
    }
}

// ---------------------------------------------------------------------------
// launcher
// ---------------------------------------------------------------------------
extern "C" void kernel_launch(void* const* d_in, const int* in_sizes, int n_in,
                              void* d_out, int out_size, void* d_ws, size_t ws_size,
                              hipStream_t stream) {
    const float* x     = (const float*)d_in[0];                  // [N,H]
    const int*   src   = (const int*)d_in[1];                    // [2,E] row0
    const int*   dst   = ((const int*)d_in[1]) + EE;             // row1
    const float* W1    = (const float*)d_in[2];
    const float* b1    = (const float*)d_in[3];
    const float* W2    = (const float*)d_in[4];
    const float* b2    = (const float*)d_in[5];
    const float* dec_w = (const float*)d_in[6];                  // [V,H,H]

    float* out     = (float*)d_out;                              // z then V*N*N
    float* z_out   = out;
    float* adj_out = out + (size_t)NN * HH;

    // workspace layout (256B aligned slabs)
    char* ws = (char*)d_ws;
    auto take = [&](size_t bytes) {
        char* p = ws;
        ws += (bytes + 255) & ~(size_t)255;
        return p;
    };
    float*     deg  = (float*)take((size_t)NN * 4);
    float*     dinv = (float*)take((size_t)NN * 4);
    float*     hbuf = (float*)take((size_t)NN * HH * 4);    // GEMM f32 out
    float*     agg  = (float*)take((size_t)NN * HH * 4);    // aggregation buf
    _Float16*  xh   = (_Float16*)take((size_t)NN * HH * 2);
    _Float16*  zh   = (_Float16*)take((size_t)NN * HH * 2); // z1 then z (f16)
    _Float16*  ph   = (_Float16*)take((size_t)NN * HH * 2); // P (f16)
    _Float16*  wt   = (_Float16*)take((size_t)HH * HH * 2); // weight^T (f16)

    const int T = 256;
    const int NH = NN * HH;

    // --- degree / dinv ---
    k_deg_init <<<NN / T, T, 0, stream>>>(deg, NN);
    k_deg_count<<<1024, T, 0, stream>>>(dst, deg, EE);
    k_dinv     <<<NN / T, T, 0, stream>>>(deg, dinv, NN);

    // --- layer 1: h = x @ W1 ; z1 = relu(Aggregate(h) + b1) ---
    k_cvt_f32_f16 <<<4096, T, 0, stream>>>(x, xh, NH);
    k_cvt_transpose<<<HH * HH / T, T, 0, stream>>>(W1, wt, HH, HH);
    k_gemm_bt_wmma<false><<<dim3(HH / 128, NN / 128), T, 0, stream>>>(
        xh, wt, hbuf, NN, HH, HH);
    k_agg_init <<<4096, T, 0, stream>>>(hbuf, dinv, agg, NH);
    k_agg_edges<<<4096, T, 0, stream>>>(hbuf, dinv, src, dst, agg, EE);
    k_finalize <<<4096, T, 0, stream>>>(agg, b1, nullptr, zh, NH, 1);

    // --- layer 2: h = z1 @ W2 ; z = Aggregate(h) + b2 ---
    k_cvt_transpose<<<HH * HH / T, T, 0, stream>>>(W2, wt, HH, HH);
    k_gemm_bt_wmma<false><<<dim3(HH / 128, NN / 128), T, 0, stream>>>(
        zh, wt, hbuf, NN, HH, HH);
    k_agg_init <<<4096, T, 0, stream>>>(hbuf, dinv, agg, NH);
    k_agg_edges<<<4096, T, 0, stream>>>(hbuf, dinv, src, dst, agg, EE);
    k_finalize <<<4096, T, 0, stream>>>(agg, b2, z_out, zh, NH, 0);

    // --- decoder per view: sigmoid((z @ w_v) @ z^T) ---
    for (int v = 0; v < VV; ++v) {
        k_cvt_transpose<<<HH * HH / T, T, 0, stream>>>(
            dec_w + (size_t)v * HH * HH, wt, HH, HH);
        k_gemm_bt_wmma<false><<<dim3(HH / 128, NN / 128), T, 0, stream>>>(
            zh, wt, hbuf, NN, HH, HH);                       // P = z @ w_v
        k_cvt_f32_f16<<<4096, T, 0, stream>>>(hbuf, ph, NH);
        k_gemm_bt_wmma<true><<<dim3(NN / 128, NN / 128), T, 0, stream>>>(
            ph, zh, adj_out + (size_t)v * NN * NN, NN, NN, HH); // sigmoid(P @ z^T)
    }
}
